// RWKVBlock_86474871538073
// MI455X (gfx1250) — matrix operand order repaired and checked by reading, not compiled, not run on hardware
//
#include <hip/hip_runtime.h>

// ---------------- problem constants ----------------
#define BB   4
#define TT   4096
#define CC   1024
#define HH   4096
#define MR   (BB * TT)          // 16384 rows

// ---------------- vector types ----------------
typedef __bf16       v16bf __attribute__((ext_vector_type(16)));
typedef float        v8f   __attribute__((ext_vector_type(8)));
typedef unsigned int u32x4 __attribute__((ext_vector_type(4)));
typedef int          i32x4 __attribute__((ext_vector_type(4)));
typedef int          i32x8 __attribute__((ext_vector_type(8)));

// ---------------- helpers ----------------
__device__ __forceinline__ unsigned short f2bf(float f) {
  unsigned u = __float_as_uint(f);
  unsigned r = u + 0x7FFFu + ((u >> 16) & 1u);   // round-to-nearest-even
  return (unsigned short)(r >> 16);
}

__device__ __forceinline__ float sigmoidf_fast(float x) {
  return 1.0f / (1.0f + __expf(-x));
}

// Block reduction of (sum, sumsq) across 256 threads (8 wave32 waves).
__device__ __forceinline__ void blk_red2(float& a, float& b, float* sm) {
#pragma unroll
  for (int off = 16; off > 0; off >>= 1) {
    a += __shfl_xor(a, off, 32);
    b += __shfl_xor(b, off, 32);
  }
  const int w = threadIdx.x >> 5;
  __syncthreads();
  if ((threadIdx.x & 31) == 0) { sm[w] = a; sm[8 + w] = b; }
  __syncthreads();
  a = 0.f; b = 0.f;
#pragma unroll
  for (int i = 0; i < 8; ++i) { a += sm[i]; b += sm[8 + i]; }
}

// ---------------- Tensor Data Mover: 2D tile (bf16) global -> LDS ----------------
// Builds a D# per CDNA5 ISA ch.8: group0 = {count|flags, lds_addr, global_addr, type},
// group1 = {data_size, tensor dims == tile dims (no OOB by construction), stride}.
__device__ __forceinline__ void tdm_load_tile_bf16(unsigned lds_byte_off,
                                                   const void* gsrc,
                                                   unsigned tile_k,       // elems per row
                                                   unsigned tile_rows,    // rows
                                                   unsigned row_stride)   // elems
{
  unsigned long long ga = (unsigned long long)(uintptr_t)gsrc;
  u32x4 g0 = { 1u,                                   // count=1, user descriptor
               lds_byte_off,                         // lds_addr
               (unsigned)(ga & 0xFFFFFFFFu),         // global_addr[31:0]
               (unsigned)((ga >> 32) & 0x1FFFFFFu) | (2u << 30) }; // addr[56:32] | type=2
  i32x8 g1 = { (int)(1u << 16),                                         // data_size=1 (2B), wg_mask=0
               (int)((tile_k & 0xFFFFu) << 16),                         // tensor_dim0 lo16
               (int)(((tile_k >> 16) & 0xFFFFu) | ((tile_rows & 0xFFFFu) << 16)), // dim0 hi | dim1 lo
               (int)(((tile_rows >> 16) & 0xFFFFu) | ((tile_k & 0xFFFFu) << 16)), // dim1 hi | tile_dim0
               (int)(tile_rows & 0xFFFFu),                              // tile_dim1 (tile_dim2=0)
               (int)row_stride,                                          // tensor_dim0_stride lo32
               0, 0 };
  i32x4 z4 = { 0, 0, 0, 0 };
#if defined(__clang_major__) && (__clang_major__ >= 23)
  i32x8 z8 = { 0, 0, 0, 0, 0, 0, 0, 0 };
  __builtin_amdgcn_tensor_load_to_lds(g0, g1, z4, z4, z8, 0);
#else
  __builtin_amdgcn_tensor_load_to_lds(g0, g1, z4, z4, 0);
#endif
}

// ---------------- WMMA fragment loads from LDS (per ISA 7.12.2 layouts) ----------------
// LDS tile: row-major, KTILE(=64) bf16 per row.
#define KTILE 64

__device__ __forceinline__ v16bf load_frag_a(const unsigned short* tile, int mrow,
                                             int kk, int lane) {
  const int r  = mrow + (lane & 15);
  const int kh = lane >> 4;
  const unsigned short* p = tile + r * KTILE + kk * 32 + kh * 8;
  union { u32x4 q[2]; v16bf v; } u;
  u.q[0] = *reinterpret_cast<const u32x4*>(p);        // K[kh*8 .. kh*8+7]
  u.q[1] = *reinterpret_cast<const u32x4*>(p + 16);   // K[kh*8+16 .. kh*8+23]
  return u.v;
}

__device__ __forceinline__ v16bf load_frag_b(const unsigned short* tile, int ncol,
                                             int kk, int lane) {
  const int r  = ncol + (lane & 15);
  const int kh = lane >> 4;
  const unsigned short* p = tile + r * KTILE + kk * 32 + kh * 8;
  union { u32x4 q[2]; v16bf v; } u;
  u.q[0] = *reinterpret_cast<const u32x4*>(p);        // K[kh*8 .. kh*8+7]
  u.q[1] = *reinterpret_cast<const u32x4*>(p + 16);   // K[kh*8+16 .. kh*8+23]
  return u.v;
}

__device__ __forceinline__ v8f wmma_bf16(v16bf a, v16bf b, v8f c) {
  return __builtin_amdgcn_wmma_f32_16x16x32_bf16(false, a, false, b,
                                                 (short)0, c, false, false);
}

// ---------------- GEMM: out[M,N] = A[M,K](bf16) * W[N,K](bf16)^T ----------------
// Block 256 thr (8 waves), 256x128 block tile, wave tile 64x64 (4x4 WMMA tiles,
// 128 f32 accum VGPRs; 16 ds_load_b128 feed 16 WMMAs per 32-deep K slice).
// A/B tiles double-buffered in 96 KB dynamic LDS, filled by the Tensor Data Mover
// (wave 0 issues the DMA, s_wait_tensorcnt + split barriers synchronize).
// EPI: 0 = f32 store, 1 = bf16(relu^2), 2 = f32 aux+acc (residual),
//      3 = f32 aux2 + sigmoid(aux)*acc (final fuse).
#define A_TILE_ELEMS (256 * KTILE)               // 16384 bf16 = 32 KB
#define B_TILE_ELEMS (128 * KTILE)               //  8192 bf16 = 16 KB
#define GEMM_LDS_BYTES ((2 * A_TILE_ELEMS + 2 * B_TILE_ELEMS) * 2)   // 96 KB

template <int EPI>
__global__ __launch_bounds__(256) void gemm_wmma(
    const unsigned short* __restrict__ A, const unsigned short* __restrict__ Wt,
    int M, int N, int K,
    float* __restrict__ outF, unsigned short* __restrict__ outB,
    const float* __restrict__ aux, const float* __restrict__ aux2)
{
  extern __shared__ unsigned short smem[];
  // NOTE: scalar pointers only — arrays of pointers into extern-shared get
  // folded into static initializers with addrspacecast, which ld.lld rejects.
  unsigned short* tA0 = smem;
  unsigned short* tA1 = smem + A_TILE_ELEMS;
  unsigned short* tB0 = smem + 2 * A_TILE_ELEMS;
  unsigned short* tB1 = smem + 2 * A_TILE_ELEMS + B_TILE_ELEMS;

  const int tid  = threadIdx.x;
  const int lane = tid & 31;
  const int wv   = tid >> 5;
  const int waveM = (wv >> 1) * 64;   // 4 waves along M
  const int waveN = (wv & 1) * 64;    // 2 waves along N
  const int row0 = blockIdx.y * 256;
  const int col0 = blockIdx.x * 128;

  v8f acc[4][4] = {};

  // LDS byte offset of the dynamic-shared base (low 32 bits of generic addr).
  const unsigned ldsBase = (unsigned)(size_t)(void*)smem;

  auto issue = [&](int ks, int buf) {
    const unsigned short* ga = A  + (size_t)row0 * K + (size_t)ks * KTILE;
    const unsigned short* gb = Wt + (size_t)col0 * K + (size_t)ks * KTILE;
    const unsigned offA = ldsBase + (buf ? (unsigned)(A_TILE_ELEMS * 2) : 0u);
    const unsigned offB = ldsBase + (unsigned)(2 * A_TILE_ELEMS * 2) +
                          (buf ? (unsigned)(B_TILE_ELEMS * 2) : 0u);
    tdm_load_tile_bf16(offA, ga, KTILE, 256, (unsigned)K);
    tdm_load_tile_bf16(offB, gb, KTILE, 128, (unsigned)K);
  };

  const int nk = K / KTILE;
  if (wv == 0) issue(0, 0);                 // prologue DMA (TENSORcnt = 2)

  for (int ks = 0; ks < nk; ++ks) {
    const int cur = ks & 1;
    if (wv == 0) {
      if (ks + 1 < nk) {
        issue(ks + 1, cur ^ 1);             // prefetch next stage (cnt <= 4)
        __builtin_amdgcn_s_wait_tensorcnt(2);   // current stage's 2 DMAs done
      } else {
        __builtin_amdgcn_s_wait_tensorcnt(0);
      }
    }
    __syncthreads();                        // LDS tile visible to all waves

    const unsigned short* pa = cur ? tA1 : tA0;
    const unsigned short* pb = cur ? tB1 : tB0;
#pragma unroll
    for (int kk = 0; kk < KTILE / 32; ++kk) {
      v16bf af[4], bfr[4];
#pragma unroll
      for (int i = 0; i < 4; ++i) af[i]  = load_frag_a(pa, waveM + i * 16, kk, lane);
#pragma unroll
      for (int j = 0; j < 4; ++j) bfr[j] = load_frag_b(pb, waveN + j * 16, kk, lane);
#pragma unroll
      for (int i = 0; i < 4; ++i)
#pragma unroll
        for (int j = 0; j < 4; ++j)
          acc[i][j] = wmma_bf16(af[i], bfr[j], acc[i][j]);
    }
    __syncthreads();                        // all reads done before buffer reuse
  }

  // Epilogue: C/D layout — VGPR r: lanes0-15 M=r, lanes16-31 M=r+8; N = lane&15.
#pragma unroll
  for (int i = 0; i < 4; ++i)
#pragma unroll
    for (int j = 0; j < 4; ++j) {
      const int mBase = row0 + waveM + i * 16 + ((lane >> 4) << 3);
      const int n     = col0 + waveN + j * 16 + (lane & 15);
#pragma unroll
      for (int r = 0; r < 8; ++r) {
        const size_t idx = (size_t)(mBase + r) * (size_t)N + (size_t)n;
        const float a = acc[i][j][r];
        if (EPI == 0) {
          outF[idx] = a;
        } else if (EPI == 1) {
          const float t = a > 0.f ? a : 0.f;
          outB[idx] = f2bf(t * t);
        } else if (EPI == 2) {
          outF[idx] = aux[idx] + a;
        } else {
          outF[idx] = aux2[idx] + a * sigmoidf_fast(aux[idx]);
        }
      }
    }
}

// ---------------- prep1: ln0 + ln1 + time-shift + lerp mix -> bf16 ----------------
__global__ __launch_bounds__(256) void prep1_kernel(
    const float* __restrict__ x,
    const float* __restrict__ ln0w, const float* __restrict__ ln0b,
    const float* __restrict__ ln1w, const float* __restrict__ ln1b,
    const float* __restrict__ tmk,  const float* __restrict__ tmv,
    const float* __restrict__ tmr,
    float* __restrict__ xbuf,
    unsigned short* __restrict__ mk, unsigned short* __restrict__ mv,
    unsigned short* __restrict__ mr)
{
  __shared__ float sm[16];
  const int row = blockIdx.x;            // b*T + t
  const int t   = row & (TT - 1);
  const int tid = threadIdx.x;
  const float invC = 1.0f / (float)CC;

  float cur[4], h[4], hp[4];
  const float* xr = x + (size_t)row * CC;
#pragma unroll
  for (int j = 0; j < 4; ++j) cur[j] = xr[tid + 256 * j];

  // ln0 of current row
  float s = 0.f, s2 = 0.f;
#pragma unroll
  for (int j = 0; j < 4; ++j) { s += cur[j]; s2 += cur[j] * cur[j]; }
  blk_red2(s, s2, sm);
  float mean = s * invC, var = s2 * invC - mean * mean;
  float rstd = rsqrtf(var + 1e-5f);
#pragma unroll
  for (int j = 0; j < 4; ++j) {
    const int c = tid + 256 * j;
    const float y = (cur[j] - mean) * rstd * ln0w[c] + ln0b[c];
    xbuf[(size_t)row * CC + c] = y;      // residual stream
    cur[j] = y;
  }
  // ln1 of ln0 output
  s = 0.f; s2 = 0.f;
#pragma unroll
  for (int j = 0; j < 4; ++j) { s += cur[j]; s2 += cur[j] * cur[j]; }
  blk_red2(s, s2, sm);
  mean = s * invC; var = s2 * invC - mean * mean; rstd = rsqrtf(var + 1e-5f);
#pragma unroll
  for (int j = 0; j < 4; ++j) {
    const int c = tid + 256 * j;
    h[j] = (cur[j] - mean) * rstd * ln1w[c] + ln1b[c];
    hp[j] = 0.f;
  }

  if (t > 0) {                            // recompute previous row (block-uniform)
    const float* xp = x + (size_t)(row - 1) * CC;
    float pv[4];
#pragma unroll
    for (int j = 0; j < 4; ++j) pv[j] = xp[tid + 256 * j];
    s = 0.f; s2 = 0.f;
#pragma unroll
    for (int j = 0; j < 4; ++j) { s += pv[j]; s2 += pv[j] * pv[j]; }
    blk_red2(s, s2, sm);
    mean = s * invC; var = s2 * invC - mean * mean; rstd = rsqrtf(var + 1e-5f);
#pragma unroll
    for (int j = 0; j < 4; ++j) {
      const int c = tid + 256 * j;
      pv[j] = (pv[j] - mean) * rstd * ln0w[c] + ln0b[c];
    }
    s = 0.f; s2 = 0.f;
#pragma unroll
    for (int j = 0; j < 4; ++j) { s += pv[j]; s2 += pv[j] * pv[j]; }
    blk_red2(s, s2, sm);
    mean = s * invC; var = s2 * invC - mean * mean; rstd = rsqrtf(var + 1e-5f);
#pragma unroll
    for (int j = 0; j < 4; ++j) {
      const int c = tid + 256 * j;
      hp[j] = (pv[j] - mean) * rstd * ln1w[c] + ln1b[c];
    }
  }

#pragma unroll
  for (int j = 0; j < 4; ++j) {
    const int c = tid + 256 * j;
    const size_t o = (size_t)row * CC + c;
    const float ak = tmk[c], av = tmv[c], ar = tmr[c];
    mk[o] = f2bf(h[j] * ak + hp[j] * (1.f - ak));
    mv[o] = f2bf(h[j] * av + hp[j] * (1.f - av));
    mr[o] = f2bf(h[j] * ar + hp[j] * (1.f - ar));
  }
}

// ---------------- prep2: ln2 + time-shift + mix -> bf16 ----------------
__global__ __launch_bounds__(256) void prep2_kernel(
    const float* __restrict__ x2,
    const float* __restrict__ ln2w, const float* __restrict__ ln2b,
    const float* __restrict__ cmk,  const float* __restrict__ cmr,
    unsigned short* __restrict__ mk2, unsigned short* __restrict__ mr2)
{
  __shared__ float sm[16];
  const int row = blockIdx.x;
  const int t   = row & (TT - 1);
  const int tid = threadIdx.x;
  const float invC = 1.0f / (float)CC;

  float h[4], hp[4];
  const float* xr = x2 + (size_t)row * CC;
#pragma unroll
  for (int j = 0; j < 4; ++j) h[j] = xr[tid + 256 * j];
  float s = 0.f, s2 = 0.f;
#pragma unroll
  for (int j = 0; j < 4; ++j) { s += h[j]; s2 += h[j] * h[j]; }
  blk_red2(s, s2, sm);
  float mean = s * invC, var = s2 * invC - mean * mean;
  float rstd = rsqrtf(var + 1e-5f);
#pragma unroll
  for (int j = 0; j < 4; ++j) {
    const int c = tid + 256 * j;
    h[j] = (h[j] - mean) * rstd * ln2w[c] + ln2b[c];
    hp[j] = 0.f;
  }

  if (t > 0) {
    const float* xp = x2 + (size_t)(row - 1) * CC;
    float pv[4];
#pragma unroll
    for (int j = 0; j < 4; ++j) pv[j] = xp[tid + 256 * j];
    s = 0.f; s2 = 0.f;
#pragma unroll
    for (int j = 0; j < 4; ++j) { s += pv[j]; s2 += pv[j] * pv[j]; }
    blk_red2(s, s2, sm);
    mean = s * invC; var = s2 * invC - mean * mean; rstd = rsqrtf(var + 1e-5f);
#pragma unroll
    for (int j = 0; j < 4; ++j) {
      const int c = tid + 256 * j;
      hp[j] = (pv[j] - mean) * rstd * ln2w[c] + ln2b[c];
    }
  }

#pragma unroll
  for (int j = 0; j < 4; ++j) {
    const int c = tid + 256 * j;
    const size_t o = (size_t)row * CC + c;
    const float ak = cmk[c], ar = cmr[c];
    mk2[o] = f2bf(h[j] * ak + hp[j] * (1.f - ak));
    mr2[o] = f2bf(h[j] * ar + hp[j] * (1.f - ar));
  }
}

// ---------------- WKV scan: serial over T, parallel over B*C channels ----------------
__global__ __launch_bounds__(256) void wkv_kernel(
    const float* __restrict__ td, const float* __restrict__ tf,
    const float* __restrict__ kbuf, const float* __restrict__ vbuf,
    const float* __restrict__ rbuf, unsigned short* __restrict__ sbuf)
{
  const int gid = blockIdx.x * 256 + threadIdx.x;   // 0 .. B*C-1
  const int b = gid >> 10;
  const int c = gid & (CC - 1);
  const float w = -__expf(td[c]);
  const float u = tf[c];
  float aa = 0.f, bb = 0.f, pp = -1e38f;
  size_t i = ((size_t)b * TT) * CC + c;
  for (int t = 0; t < TT; ++t, i += CC) {
    const float kt = kbuf[i];
    const float vt = vbuf[i];
    const float ww = u + kt;
    const float p  = fmaxf(pp, ww);
    float e1 = __expf(pp - p);
    float e2 = __expf(ww - p);
    const float y = (e1 * aa + e2 * vt) / (e1 * bb + e2);
    const float ww2 = pp + w;
    const float p2  = fmaxf(ww2, kt);
    e1 = __expf(ww2 - p2);
    e2 = __expf(kt - p2);
    aa = e1 * aa + e2 * vt;
    bb = e1 * bb + e2;
    pp = p2;
    sbuf[i] = f2bf(y * sigmoidf_fast(rbuf[i]));
  }
}

// ---------------- fp32 -> bf16 weight conversion ----------------
__global__ __launch_bounds__(256) void f32_to_bf16_kernel(
    const float* __restrict__ s, unsigned short* __restrict__ d, int n)
{
  const int i = blockIdx.x * 256 + threadIdx.x;
  if (i < n) d[i] = f2bf(s[i]);
}

// ---------------- host launch ----------------
extern "C" void kernel_launch(void* const* d_in, const int* in_sizes, int n_in,
                              void* d_out, int out_size, void* d_ws, size_t ws_size,
                              hipStream_t stream)
{
  (void)in_sizes; (void)n_in; (void)out_size; (void)ws_size;
  const float* x    = (const float*)d_in[0];
  const float* ln0w = (const float*)d_in[1];
  const float* ln0b = (const float*)d_in[2];
  const float* ln1w = (const float*)d_in[3];
  const float* ln1b = (const float*)d_in[4];
  const float* ln2w = (const float*)d_in[5];
  const float* ln2b = (const float*)d_in[6];
  const float* td   = (const float*)d_in[7];
  const float* tf   = (const float*)d_in[8];
  const float* tmk  = (const float*)d_in[9];
  const float* tmv  = (const float*)d_in[10];
  const float* tmr  = (const float*)d_in[11];
  const float* Wk   = (const float*)d_in[12];
  const float* Wv   = (const float*)d_in[13];
  const float* Wr   = (const float*)d_in[14];
  const float* Wo   = (const float*)d_in[15];
  const float* cmk  = (const float*)d_in[16];
  const float* cmr  = (const float*)d_in[17];
  const float* Wfk  = (const float*)d_in[18];
  const float* Wfr  = (const float*)d_in[19];
  const float* Wfv  = (const float*)d_in[20];

  const size_t MB = 1ull << 20;
  char* w = (char*)d_ws;
  float*          xbuf = (float*)(w + 0 * MB);            // 64 MB residual stream
  float*          kbuf = (float*)(w + 64 * MB);           // 64 MB
  float*          vbuf = (float*)(w + 128 * MB);          // 64 MB
  float*          rbuf = (float*)(w + 192 * MB);          // 64 MB
  unsigned short* mk   = (unsigned short*)(w + 256 * MB); // 32 MB bf16
  unsigned short* mv   = (unsigned short*)(w + 288 * MB); // 32 MB bf16 (later sbuf)
  unsigned short* mr   = (unsigned short*)(w + 320 * MB); // 32 MB bf16
  unsigned short* wbf  = (unsigned short*)(w + 352 * MB); // 26 MB bf16 weights
  unsigned short* kf   = (unsigned short*)(w + 64 * MB);  // 128 MB bf16 (aliases k+v)

  const size_t M1 = 1024 * 1024;
  unsigned short* wk_bf  = wbf + 0 * M1;
  unsigned short* wv_bf  = wbf + 1 * M1;
  unsigned short* wr_bf  = wbf + 2 * M1;
  unsigned short* wo_bf  = wbf + 3 * M1;
  unsigned short* wfk_bf = wbf + 4 * M1;   // 4M elems
  unsigned short* wfr_bf = wbf + 8 * M1;   // 1M elems
  unsigned short* wfv_bf = wbf + 9 * M1;   // 4M elems

  // 1) convert weights to bf16 (recomputed every call; deterministic)
  f32_to_bf16_kernel<<<(int)(M1 / 256), 256, 0, stream>>>(Wk,  wk_bf,  (int)M1);
  f32_to_bf16_kernel<<<(int)(M1 / 256), 256, 0, stream>>>(Wv,  wv_bf,  (int)M1);
  f32_to_bf16_kernel<<<(int)(M1 / 256), 256, 0, stream>>>(Wr,  wr_bf,  (int)M1);
  f32_to_bf16_kernel<<<(int)(M1 / 256), 256, 0, stream>>>(Wo,  wo_bf,  (int)M1);
  f32_to_bf16_kernel<<<(int)(4 * M1 / 256), 256, 0, stream>>>(Wfk, wfk_bf, (int)(4 * M1));
  f32_to_bf16_kernel<<<(int)(M1 / 256), 256, 0, stream>>>(Wfr, wfr_bf, (int)M1);
  f32_to_bf16_kernel<<<(int)(4 * M1 / 256), 256, 0, stream>>>(Wfv, wfv_bf, (int)(4 * M1));

  // 2) ln0 + ln1 + time-shift + mixes
  prep1_kernel<<<MR, 256, 0, stream>>>(x, ln0w, ln0b, ln1w, ln1b, tmk, tmv, tmr,
                                       xbuf, mk, mv, mr);

  // 3) k / v / r projections (WMMA, TDM-staged); 256x128 block tiles
  dim3 gC(CC / 128, MR / 256);      // (8, 64)
  dim3 gH(HH / 128, MR / 256);      // (32, 64)
  const unsigned shmem = GEMM_LDS_BYTES;   // 96 KB dynamic LDS
  gemm_wmma<0><<<gC, 256, shmem, stream>>>(mk, wk_bf, MR, CC, CC, kbuf, nullptr, nullptr, nullptr);
  gemm_wmma<0><<<gC, 256, shmem, stream>>>(mv, wv_bf, MR, CC, CC, vbuf, nullptr, nullptr, nullptr);
  gemm_wmma<0><<<gC, 256, shmem, stream>>>(mr, wr_bf, MR, CC, CC, rbuf, nullptr, nullptr, nullptr);

  // 4) WKV scan + sigmoid(r)*y -> bf16 (reuses mv storage)
  wkv_kernel<<<(BB * CC) / 256, 256, 0, stream>>>(td, tf, kbuf, vbuf, rbuf, mv);

  // 5) output projection with fused residual: x = x + (sr*att) @ Wo^T
  gemm_wmma<2><<<gC, 256, shmem, stream>>>(mv, wo_bf, MR, CC, CC, xbuf, nullptr, xbuf, nullptr);

  // 6) channel-mix prep
  prep2_kernel<<<MR, 256, 0, stream>>>(xbuf, ln2w, ln2b, cmk, cmr, mk, mr);

  // 7) r2 logits
  gemm_wmma<0><<<gC, 256, shmem, stream>>>(mr, wfr_bf, MR, CC, CC, rbuf, nullptr, nullptr, nullptr);

  // 8) kf = bf16(relu(xk2 @ Wfk^T)^2)   [N = 4096]
  gemm_wmma<1><<<gH, 256, shmem, stream>>>(mk, wfk_bf, MR, HH, CC, nullptr, kf, nullptr, nullptr);

  // 9) out = x + sigmoid(r2) * (kf @ Wfv^T)   [K = 4096]
  gemm_wmma<3><<<gC, 256, shmem, stream>>>(kf, wfv_bf, MR, CC, HH, (float*)d_out, nullptr,
                                           rbuf, xbuf);
}
